// GroupInteraction_82188494176307
// MI455X (gfx1250) — matrix operand order repaired and checked by reading, not compiled, not run on hardware
//
#include <hip/hip_runtime.h>
#include <hip/hip_bf16.h>

typedef __attribute__((ext_vector_type(16))) _Float16 v16h;
typedef __attribute__((ext_vector_type(8)))  _Float16 v8h;
typedef __attribute__((ext_vector_type(8)))  float    v8f;

#define BB 32
#define SS 512
#define EE 512
#define HH 8
#define DD 64

// ---------------------------------------------------------------------------
// Fragment helpers (CDNA5 wave32 WMMA 16x16x32 f16 layouts, ISA 7.12.2)
// ---------------------------------------------------------------------------

// A fragment from row-major [M][K] f16, tile row m0, k offset k0, row stride lda.
__device__ inline v16h load_a_frag_f16(const _Float16* __restrict__ A, int lda,
                                       int m0, int k0, int lane) {
    const int g = lane >> 4, m = lane & 15;
    const _Float16* p = A + (size_t)(m0 + m) * lda + k0 + 8 * g;
    v8h lo = *(const v8h*)p;            // K = k0+8g .. +7
    v8h hi = *(const v8h*)(p + 16);     // K = k0+16+8g .. +7
    v16h r;
#pragma unroll
    for (int i = 0; i < 8; ++i) { r[i] = lo[i]; r[8 + i] = hi[i]; }
    return r;
}

// A fragment from row-major [M][K] fp32 (converted to f16 on the fly).
__device__ inline v16h load_a_frag_f32(const float* __restrict__ A, int lda,
                                       int m0, int k0, int lane) {
    const int g = lane >> 4, m = lane & 15;
    const float* p = A + (size_t)(m0 + m) * lda + k0 + 8 * g;
    v8f lo = *(const v8f*)p;
    v8f hi = *(const v8f*)(p + 16);
    v16h r;
#pragma unroll
    for (int i = 0; i < 8; ++i) { r[i] = (_Float16)lo[i]; r[8 + i] = (_Float16)hi[i]; }
    return r;
}

// B fragment from B-transposed row-major [N][K] f16, tile col n0, k offset k0.
__device__ inline v16h load_b_frag(const _Float16* __restrict__ Bt, int ldb,
                                   int n0, int k0, int lane) {
    const int g = lane >> 4, n = lane & 15;
    const _Float16* p = Bt + (size_t)(n0 + n) * ldb + k0 + 8 * g;
    v8h lo = *(const v8h*)p;            // K = k0+8g .. +7
    v8h hi = *(const v8h*)(p + 16);     // K = k0+16+8g .. +7
    v16h r;
#pragma unroll
    for (int i = 0; i < 8; ++i) { r[i] = lo[i]; r[8 + i] = hi[i]; }
    return r;
}

__device__ inline v8f wmma_f16(v16h a, v16h b, v8f c) {
    return __builtin_amdgcn_wmma_f32_16x16x32_f16(false, a, false, b,
                                                  (short)0, c, false, false);
}

// ---------------------------------------------------------------------------
// fp32 -> f16 conversion
// ---------------------------------------------------------------------------
__global__ void __launch_bounds__(256)
cvt_f32_to_f16(const float* __restrict__ src, _Float16* __restrict__ dst,
               long long n) {
    long long i = (long long)blockIdx.x * blockDim.x + threadIdx.x;
    long long stride = (long long)gridDim.x * blockDim.x;
    for (; i < n; i += stride) dst[i] = (_Float16)src[i];
}

// ---------------------------------------------------------------------------
// Stage 1: qkv = x @ w_in^T + b_in ; scatter into Q[b,h,s,d], K[b,h,s,d],
// Vt[b,h,d,s] (all f16). M=16384, N=1536, K=512. Wave tile: 32x64.
// ---------------------------------------------------------------------------
__global__ void __launch_bounds__(256)
qkv_gemm_kernel(const _Float16* __restrict__ xh, const _Float16* __restrict__ wih,
                const float* __restrict__ b_in, _Float16* __restrict__ qh,
                _Float16* __restrict__ kh, _Float16* __restrict__ vt) {
    const int lane = threadIdx.x & 31;
    const int wid  = blockIdx.x * 8 + (threadIdx.x >> 5);
    const int NB   = (3 * EE) / 64;               // 24 n-strips
    const int mt = wid / NB, nb = wid % NB;
    const int m0 = mt * 32, n0 = nb * 64;
    const int g = lane >> 4, c = lane & 15;

    v8f acc[2][4];
#pragma unroll
    for (int t = 0; t < 2; ++t)
#pragma unroll
        for (int j = 0; j < 4; ++j) acc[t][j] = (v8f)(0.0f);

    for (int k0 = 0; k0 < EE; k0 += 32) {
        v16h a0 = load_a_frag_f16(xh, EE, m0,      k0, lane);
        v16h a1 = load_a_frag_f16(xh, EE, m0 + 16, k0, lane);
#pragma unroll
        for (int j = 0; j < 4; ++j) {
            v16h b = load_b_frag(wih, EE, n0 + j * 16, k0, lane);
            acc[0][j] = wmma_f16(a0, b, acc[0][j]);
            acc[1][j] = wmma_f16(a1, b, acc[1][j]);
        }
    }

#pragma unroll
    for (int j = 0; j < 4; ++j) {
        const int n   = n0 + j * 16 + c;           // global output feature
        const int sec = n >> 9;                    // 0=Q 1=K 2=V (uniform per j)
        const int e   = n & 511;
        const int h   = e >> 6, d = e & 63;
        const float bias = b_in[n];
#pragma unroll
        for (int t = 0; t < 2; ++t)
#pragma unroll
            for (int i = 0; i < 8; ++i) {
                const int m  = m0 + 16 * t + 8 * g + i;
                const int bb = m >> 9, s = m & 511;
                const _Float16 hv = (_Float16)(acc[t][j][i] + bias);
                const size_t bh = (size_t)bb * HH + h;
                if (sec == 0)      qh[(bh * SS + s) * DD + d] = hv;
                else if (sec == 1) kh[(bh * SS + s) * DD + d] = hv;
                else               vt[(bh * DD + d) * SS + s] = hv;
            }
    }
}

// ---------------------------------------------------------------------------
// Stage 2: scores[b,h,q,k] = (Q @ K^T) / 8. Per (b,h): M=N=512, K=64.
// Wave tile: 32x64.
// ---------------------------------------------------------------------------
__global__ void __launch_bounds__(256)
scores_kernel(const _Float16* __restrict__ qh, const _Float16* __restrict__ kh,
              float* __restrict__ att) {
    const int lane = threadIdx.x & 31;
    const int wid  = blockIdx.x * 8 + (threadIdx.x >> 5);
    const int nb = wid & 7;               // 8 strips of 64 keys
    const int mt = (wid >> 3) & 15;       // 16 query tiles of 32
    const int bh = wid >> 7;              // 256 (b,h) pairs
    const int m0 = mt * 32, n0 = nb * 64;
    const int g = lane >> 4, c = lane & 15;

    const _Float16* Q = qh + (size_t)bh * SS * DD;
    const _Float16* K = kh + (size_t)bh * SS * DD;
    float* A = att + (size_t)bh * SS * SS;

    v8f acc[2][4];
#pragma unroll
    for (int t = 0; t < 2; ++t)
#pragma unroll
        for (int j = 0; j < 4; ++j) acc[t][j] = (v8f)(0.0f);

#pragma unroll
    for (int k0 = 0; k0 < DD; k0 += 32) {
        v16h a0 = load_a_frag_f16(Q, DD, m0,      k0, lane);
        v16h a1 = load_a_frag_f16(Q, DD, m0 + 16, k0, lane);
#pragma unroll
        for (int j = 0; j < 4; ++j) {
            v16h b = load_b_frag(K, DD, n0 + j * 16, k0, lane);
            acc[0][j] = wmma_f16(a0, b, acc[0][j]);
            acc[1][j] = wmma_f16(a1, b, acc[1][j]);
        }
    }

#pragma unroll
    for (int t = 0; t < 2; ++t)
#pragma unroll
        for (int j = 0; j < 4; ++j)
#pragma unroll
            for (int i = 0; i < 8; ++i)
                A[(size_t)(m0 + 16 * t + 8 * g + i) * SS + n0 + j * 16 + c] =
                    acc[t][j][i] * 0.125f;
}

// ---------------------------------------------------------------------------
// Stage 3: row softmax over the last dim of att (in place). One block per row.
// ---------------------------------------------------------------------------
__global__ void __launch_bounds__(256)
softmax_kernel(float* __restrict__ att) {
    __shared__ float red[256];
    float* p = att + (size_t)blockIdx.x * SS;
    const int t = threadIdx.x;
    float v0 = p[t], v1 = p[t + 256];

    red[t] = fmaxf(v0, v1);
    __syncthreads();
    for (int s = 128; s > 0; s >>= 1) {
        if (t < s) red[t] = fmaxf(red[t], red[t + s]);
        __syncthreads();
    }
    const float m = red[0];
    __syncthreads();

    const float e0 = __expf(v0 - m), e1 = __expf(v1 - m);
    red[t] = e0 + e1;
    __syncthreads();
    for (int s = 128; s > 0; s >>= 1) {
        if (t < s) red[t] += red[t + s];
        __syncthreads();
    }
    const float inv = 1.0f / red[0];
    p[t] = e0 * inv;
    p[t + 256] = e1 * inv;
}

// ---------------------------------------------------------------------------
// Stage 4: z[b,s,h*64+d] = att @ V. Per (b,h): M=512, N=64, K=512.
// A read from fp32 att (cvt on the fly), B from Vt[d][s]. Wave tile: 32x64.
// ---------------------------------------------------------------------------
__global__ void __launch_bounds__(256)
attv_kernel(const float* __restrict__ att, const _Float16* __restrict__ vt,
            _Float16* __restrict__ zh) {
    const int lane = threadIdx.x & 31;
    const int wid  = blockIdx.x * 8 + (threadIdx.x >> 5);
    const int mt = wid & 15;              // 16 query tiles of 32
    const int bh = wid >> 4;              // 256 (b,h) pairs
    const int bb = bh >> 3, h = bh & 7;
    const int m0 = mt * 32;
    const int g = lane >> 4, c = lane & 15;

    const float* A = att + (size_t)bh * SS * SS;
    const _Float16* V = vt + (size_t)bh * DD * SS;   // [d][s]

    v8f acc[2][4];
#pragma unroll
    for (int t = 0; t < 2; ++t)
#pragma unroll
        for (int j = 0; j < 4; ++j) acc[t][j] = (v8f)(0.0f);

    for (int k0 = 0; k0 < SS; k0 += 32) {
        v16h a0 = load_a_frag_f32(A, SS, m0,      k0, lane);
        v16h a1 = load_a_frag_f32(A, SS, m0 + 16, k0, lane);
#pragma unroll
        for (int j = 0; j < 4; ++j) {
            v16h b = load_b_frag(V, SS, j * 16, k0, lane);
            acc[0][j] = wmma_f16(a0, b, acc[0][j]);
            acc[1][j] = wmma_f16(a1, b, acc[1][j]);
        }
    }

#pragma unroll
    for (int t = 0; t < 2; ++t)
#pragma unroll
        for (int j = 0; j < 4; ++j)
#pragma unroll
            for (int i = 0; i < 8; ++i) {
                const int s = m0 + 16 * t + 8 * g + i;
                const int d = j * 16 + c;
                zh[((size_t)bb * SS + s) * EE + h * DD + d] = (_Float16)acc[t][j][i];
            }
}

// ---------------------------------------------------------------------------
// Stage 5a: t = z @ w_out^T + b_out + x (residual). M=16384, N=512, K=512.
// Wave tile: 32x64.
// ---------------------------------------------------------------------------
__global__ void __launch_bounds__(256)
outproj_kernel(const _Float16* __restrict__ zh, const _Float16* __restrict__ woh,
               const float* __restrict__ b_out, const float* __restrict__ x,
               float* __restrict__ t) {
    const int lane = threadIdx.x & 31;
    const int wid  = blockIdx.x * 8 + (threadIdx.x >> 5);
    const int nb = wid & 7, mt = wid >> 3;
    const int m0 = mt * 32, n0 = nb * 64;
    const int g = lane >> 4, c = lane & 15;

    v8f acc[2][4];
#pragma unroll
    for (int u = 0; u < 2; ++u)
#pragma unroll
        for (int j = 0; j < 4; ++j) acc[u][j] = (v8f)(0.0f);

    for (int k0 = 0; k0 < EE; k0 += 32) {
        v16h a0 = load_a_frag_f16(zh, EE, m0,      k0, lane);
        v16h a1 = load_a_frag_f16(zh, EE, m0 + 16, k0, lane);
#pragma unroll
        for (int j = 0; j < 4; ++j) {
            v16h b = load_b_frag(woh, EE, n0 + j * 16, k0, lane);
            acc[0][j] = wmma_f16(a0, b, acc[0][j]);
            acc[1][j] = wmma_f16(a1, b, acc[1][j]);
        }
    }

#pragma unroll
    for (int j = 0; j < 4; ++j) {
        const int n = n0 + j * 16 + c;
        const float bias = b_out[n];
#pragma unroll
        for (int u = 0; u < 2; ++u)
#pragma unroll
            for (int i = 0; i < 8; ++i) {
                const size_t m = m0 + 16 * u + 8 * g + i;
                t[m * EE + n] = acc[u][j][i] + bias + x[m * EE + n];
            }
    }
}

// ---------------------------------------------------------------------------
// Stage 5b: LayerNorm rows of t -> y. One block per row of 512.
// ---------------------------------------------------------------------------
__global__ void __launch_bounds__(256)
layernorm_kernel(const float* __restrict__ t, const float* __restrict__ gamma,
                 const float* __restrict__ beta, float* __restrict__ y) {
    __shared__ float red[256];
    const float* p = t + (size_t)blockIdx.x * EE;
    const int tid = threadIdx.x;
    float v0 = p[tid], v1 = p[tid + 256];

    red[tid] = v0 + v1;
    __syncthreads();
    for (int s = 128; s > 0; s >>= 1) {
        if (tid < s) red[tid] += red[tid + s];
        __syncthreads();
    }
    const float mean = red[0] * (1.0f / EE);
    __syncthreads();

    red[tid] = v0 * v0 + v1 * v1;
    __syncthreads();
    for (int s = 128; s > 0; s >>= 1) {
        if (tid < s) red[tid] += red[tid + s];
        __syncthreads();
    }
    const float var = red[0] * (1.0f / EE) - mean * mean;
    const float rstd = rsqrtf(var + 1e-5f);

    float* o = y + (size_t)blockIdx.x * EE;
    o[tid]       = (v0 - mean) * rstd * gamma[tid]       + beta[tid];
    o[tid + 256] = (v1 - mean) * rstd * gamma[tid + 256] + beta[tid + 256];
}

// ---------------------------------------------------------------------------
extern "C" void kernel_launch(void* const* d_in, const int* in_sizes, int n_in,
                              void* d_out, int out_size, void* d_ws, size_t ws_size,
                              hipStream_t stream) {
    (void)in_sizes; (void)n_in; (void)out_size; (void)ws_size;
    const float* x     = (const float*)d_in[0];   // [B,S,E]
    const float* w_in  = (const float*)d_in[1];   // [3E,E]
    const float* b_in  = (const float*)d_in[2];   // [3E]
    const float* w_out = (const float*)d_in[3];   // [E,E]
    const float* b_out = (const float*)d_in[4];   // [E]
    const float* gamma = (const float*)d_in[5];   // [E]
    const float* beta  = (const float*)d_in[6];   // [E]

    float* y   = (float*)d_out;                               // [B,S,E]
    float* att = (float*)d_out + (size_t)BB * SS * EE;        // [B,H,S,S]

    // Workspace carve (all sizes multiples of 256 bytes)
    char* w = (char*)d_ws;
    const size_t M = (size_t)BB * SS;                 // 16384
    _Float16* xh  = (_Float16*)w; w += M * EE * 2;                      // x f16
    _Float16* wih = (_Float16*)w; w += (size_t)3 * EE * EE * 2;         // w_in f16
    _Float16* woh = (_Float16*)w; w += (size_t)EE * EE * 2;             // w_out f16
    _Float16* qh  = (_Float16*)w; w += (size_t)BB * HH * SS * DD * 2;   // Q [b,h,s,d]
    _Float16* kh  = (_Float16*)w; w += (size_t)BB * HH * SS * DD * 2;   // K [b,h,s,d]
    _Float16* vt  = (_Float16*)w; w += (size_t)BB * HH * DD * SS * 2;   // V^T [b,h,d,s]
    _Float16* zh  = (_Float16*)w; w += M * EE * 2;                      // z f16
    float*    tpre = (float*)w;  w += M * EE * 4;                       // pre-LN fp32

    // f16 conversions
    cvt_f32_to_f16<<<4096, 256, 0, stream>>>(x, xh, (long long)M * EE);
    cvt_f32_to_f16<<<1024, 256, 0, stream>>>(w_in, wih, (long long)3 * EE * EE);
    cvt_f32_to_f16<<<512, 256, 0, stream>>>(w_out, woh, (long long)EE * EE);

    // QKV projection: 512 m-tiles * 24 n-strips = 12288 waves -> 1536 blocks
    qkv_gemm_kernel<<<1536, 256, 0, stream>>>(xh, wih, b_in, qh, kh, vt);

    // Scores: 256 bh * 16 mt * 8 nb = 32768 waves -> 4096 blocks
    scores_kernel<<<4096, 256, 0, stream>>>(qh, kh, att);

    // Softmax: one block per row, B*H*S rows
    softmax_kernel<<<BB * HH * SS, 256, 0, stream>>>(att);

    // att @ V: 256 bh * 16 mt = 4096 waves -> 512 blocks
    attv_kernel<<<512, 256, 0, stream>>>(att, vt, zh);

    // Out projection + bias + residual: 512 mt * 8 nb = 4096 waves -> 512 blocks
    outproj_kernel<<<512, 256, 0, stream>>>(zh, woh, b_out, x, tpre);

    // LayerNorm rows -> y
    layernorm_kernel<<<16384, 256, 0, stream>>>(tpre, gamma, beta, y);
}
